// EM_AE_V1_66185446032104
// MI455X (gfx1250) — matrix operand order repaired and checked by reading, not compiled, not run on hardware
//
#include <hip/hip_runtime.h>
#include <hip/hip_bf16.h>

// Problem constants (match reference)
#define KK 8
#define BB 8192
#define DD 1024
#define HH 256

typedef __attribute__((ext_vector_type(16))) __bf16 v16bf;
typedef __attribute__((ext_vector_type(8)))  __bf16 v8bf;
typedef __attribute__((ext_vector_type(8)))  float  v8f;
typedef unsigned short ushort_t;

union ABu { v16bf v; v8bf h[2]; };

__device__ __forceinline__ ushort_t f2bf(float f) {
    unsigned int u = __builtin_bit_cast(unsigned int, f);
    u += 0x7FFFu + ((u >> 16) & 1u);           // round-to-nearest-even
    return (ushort_t)(u >> 16);
}

__device__ __forceinline__ v8f wmma_bf16(const v16bf& a, const v16bf& b, const v8f& c) {
    return __builtin_amdgcn_wmma_f32_16x16x32_bf16(false, a, false, b, (short)0, c, false, false);
}

// ---------------------------------------------------------------- small utils

__global__ void k_zero_counts(int* counts) {
    if (threadIdx.x < KK) counts[threadIdx.x] = 0;
}

__global__ void k_cvt_x(const float* __restrict__ x, ushort_t* __restrict__ xb) {
    int i = blockIdx.x * blockDim.x + threadIdx.x;
    if (i < BB * DD) xb[i] = f2bf(x[i]);
}

// Transpose+convert W1 [K,D,H] -> w1t [K,H,D] bf16, W2 [K,H,D] -> w2t [K,D,H] bf16.
__global__ void k_prep_w(const float* __restrict__ W1, const float* __restrict__ W2,
                         ushort_t* __restrict__ w1t, ushort_t* __restrict__ w2t) {
    int i = blockIdx.x * blockDim.x + threadIdx.x;
    if (i >= KK * DD * HH) return;
    {
        int k = i / (DD * HH);
        int rem = i - k * (DD * HH);
        int d = rem / HH;
        int h = rem - d * HH;
        w1t[(size_t)k * HH * DD + (size_t)h * DD + d] = f2bf(W1[i]);
    }
    {
        int k = i / (HH * DD);
        int rem = i - k * (HH * DD);
        int h = rem / DD;
        int d = rem - h * DD;
        w2t[(size_t)k * DD * HH + (size_t)d * HH + h] = f2bf(W2[i]);
    }
}

// ----------------------------------------------------- GEMM1: h = relu(xW1+b1)
// grid (B/32, K), block 128 (4 waves). Wave w: 32 rows x 64 cols (cols w*64..).
// B fragment reused by two row-group WMMAs: 12 loads : 8 wmma per 32-K step.
__global__ __launch_bounds__(128) void k_gemm1(
    const ushort_t* __restrict__ xb,    // [B,D] bf16
    const ushort_t* __restrict__ w1t,   // [K,H,D] bf16
    const float*    __restrict__ b1,    // [K,H]
    ushort_t*       __restrict__ hbuf)  // [K,B,H] bf16
{
    const int lane = threadIdx.x & 31;
    const int wave = threadIdx.x >> 5;          // 0..3
    const int k    = blockIdx.y;
    const int m0   = blockIdx.x * 32;
    const int half = lane >> 4;
    const int colB = lane & 15;
    const int n0   = wave * 64;

    const ushort_t* aRow0 = xb + (size_t)(m0 + colB) * DD;       // rows m0..m0+15
    const ushort_t* aRow1 = aRow0 + (size_t)16 * DD;             // rows m0+16..m0+31
    const ushort_t* wBase = w1t + (size_t)k * HH * DD;

    v8f acc0[4] = {};
    v8f acc1[4] = {};
    for (int k0 = 0; k0 < DD; k0 += 32) {
        ABu a0, a1;
        a0.h[0] = *(const v8bf*)(aRow0 + k0 + half * 16);
        a0.h[1] = *(const v8bf*)(aRow0 + k0 + 8 + half * 16);
        a1.h[0] = *(const v8bf*)(aRow1 + k0 + half * 16);
        a1.h[1] = *(const v8bf*)(aRow1 + k0 + 8 + half * 16);
#pragma unroll
        for (int t = 0; t < 4; ++t) {
            int col = n0 + t * 16 + colB;
            v16bf b = *(const v16bf*)(wBase + (size_t)col * DD + k0 + half * 16);
            acc0[t] = wmma_bf16(a0.v, b, acc0[t]);
            acc1[t] = wmma_bf16(a1.v, b, acc1[t]);
        }
    }
#pragma unroll
    for (int t = 0; t < 4; ++t) {
        int col = n0 + t * 16 + colB;
        float bias = b1[k * HH + col];
#pragma unroll
        for (int r = 0; r < 8; ++r) {
            int orow = m0 + r + half * 8;
            float v0 = acc0[t][r] + bias;
            float v1 = acc1[t][r] + bias;
            v0 = v0 > 0.0f ? v0 : 0.0f;
            v1 = v1 > 0.0f ? v1 : 0.0f;
            hbuf[((size_t)k * BB + orow) * HH + col]      = f2bf(v0);
            hbuf[((size_t)k * BB + orow + 16) * HH + col] = f2bf(v1);
        }
    }
}

// --------------------------------------- GEMM2 fused with per-row error reduce
// grid (B/256, K), block 256 (8 waves). Wave: 32 rows x all 1024 cols.
// All 16 A fragments hoisted (whole K=H=256); n-loop is ~1 load per WMMA.
__global__ __launch_bounds__(256) void k_gemm2_err(
    const ushort_t* __restrict__ hbuf,  // [K,B,H] bf16
    const ushort_t* __restrict__ w2t,   // [K,D,H] bf16
    const float*    __restrict__ b2,    // [K,D]
    const float*    __restrict__ x,     // [B,D]
    float*          __restrict__ err)   // [K,B]
{
    const int lane = threadIdx.x & 31;
    const int wave = threadIdx.x >> 5;          // 0..7
    const int k    = blockIdx.y;
    const int m0   = blockIdx.x * 256 + wave * 32;
    const int half = lane >> 4;
    const int colB = lane & 15;

    const ushort_t* aRow0 = hbuf + ((size_t)k * BB + m0 + colB) * HH;
    const ushort_t* aRow1 = aRow0 + (size_t)16 * HH;
    const ushort_t* wBase = w2t + (size_t)k * DD * HH;

    ABu a0[8], a1[8];                            // 2 x 64 VGPRs, loaded once
#pragma unroll
    for (int s = 0; s < 8; ++s) {
        int h0 = s * 32;
        a0[s].h[0] = *(const v8bf*)(aRow0 + h0 + half * 16);
        a0[s].h[1] = *(const v8bf*)(aRow0 + h0 + 8 + half * 16);
        a1[s].h[0] = *(const v8bf*)(aRow1 + h0 + half * 16);
        a1[s].h[1] = *(const v8bf*)(aRow1 + h0 + 8 + half * 16);
    }

    float errAcc0[8] = {0.f, 0.f, 0.f, 0.f, 0.f, 0.f, 0.f, 0.f};
    float errAcc1[8] = {0.f, 0.f, 0.f, 0.f, 0.f, 0.f, 0.f, 0.f};

    for (int nc = 0; nc < DD / 64; ++nc) {
        const int n0 = nc * 64;
        v8f acc0[4] = {};
        v8f acc1[4] = {};
#pragma unroll
        for (int s = 0; s < 8; ++s) {
#pragma unroll
            for (int t = 0; t < 4; ++t) {
                int col = n0 + t * 16 + colB;
                v16bf b = *(const v16bf*)(wBase + (size_t)col * HH + s * 32 + half * 16);
                acc0[t] = wmma_bf16(a0[s].v, b, acc0[t]);
                acc1[t] = wmma_bf16(a1[s].v, b, acc1[t]);
            }
        }
#pragma unroll
        for (int t = 0; t < 4; ++t) {
            int col = n0 + t * 16 + colB;
            float bias = b2[k * DD + col];
#pragma unroll
            for (int r = 0; r < 8; ++r) {
                int orow = m0 + r + half * 8;
                float d0 = acc0[t][r] + bias - x[(size_t)orow * DD + col];
                float d1 = acc1[t][r] + bias - x[(size_t)(orow + 16) * DD + col];
                errAcc0[r] += d0 * d0;
                errAcc1[r] += d1 * d1;
            }
        }
    }
    // Reduce each errAcc[r] across its 16-lane half (xor masks stay within half).
#pragma unroll
    for (int r = 0; r < 8; ++r) {
        float v0 = errAcc0[r];
        float v1 = errAcc1[r];
        v0 += __shfl_xor(v0, 1, 32);
        v1 += __shfl_xor(v1, 1, 32);
        v0 += __shfl_xor(v0, 2, 32);
        v1 += __shfl_xor(v1, 2, 32);
        v0 += __shfl_xor(v0, 4, 32);
        v1 += __shfl_xor(v1, 4, 32);
        v0 += __shfl_xor(v0, 8, 32);
        v1 += __shfl_xor(v1, 8, 32);
        if (colB == 0) {
            err[(size_t)k * BB + m0 + r + half * 8]      = v0 * (1.0f / (float)DD);
            err[(size_t)k * BB + m0 + 16 + r + half * 8] = v1 * (1.0f / (float)DD);
        }
    }
}

// ------------------------------------------------- argmin + expert compaction
__global__ void k_assign(const float* __restrict__ err, int* __restrict__ assignv,
                         int* __restrict__ counts) {
    int b = blockIdx.x * blockDim.x + threadIdx.x;
    if (b >= BB) return;
    float best = err[b];
    int bi = 0;
#pragma unroll
    for (int k = 1; k < KK; ++k) {
        float e = err[(size_t)k * BB + b];
        if (e < best) { best = e; bi = k; }
    }
    assignv[b] = bi;
    atomicAdd(&counts[bi], 1);
}

__global__ void k_scan(const int* __restrict__ counts, int* __restrict__ offsets,
                       int* __restrict__ cursors) {
    if (threadIdx.x == 0) {
        int acc = 0;
        for (int k = 0; k < KK; ++k) {
            offsets[k] = acc;
            cursors[k] = acc;
            acc += counts[k];
        }
    }
}

__global__ void k_scatter(const int* __restrict__ assignv, int* __restrict__ cursors,
                          int* __restrict__ perm) {
    int b = blockIdx.x * blockDim.x + threadIdx.x;
    if (b >= BB) return;
    int e = assignv[b];
    int pos = atomicAdd(&cursors[e], 1);
    perm[pos] = b;
}

// ------------------------- Final: recompute recon only for the winning expert
// grid (B/16, K, D/512), block 256 (8 waves x 64 cols). Gathered A rows.
__global__ __launch_bounds__(256) void k_final(
    const ushort_t* __restrict__ hbuf,  // [K,B,H] bf16
    const ushort_t* __restrict__ w2t,   // [K,D,H] bf16
    const float*    __restrict__ b2,    // [K,D]
    const int*      __restrict__ counts,
    const int*      __restrict__ offsets,
    const int*      __restrict__ perm,
    float*          __restrict__ out)   // [B,D]
{
    const int k   = blockIdx.y;
    const int cnt = counts[k];
    const int tile = blockIdx.x;
    if (tile * 16 >= cnt) return;

    const int lane = threadIdx.x & 31;
    const int wave = threadIdx.x >> 5;
    const int half = lane >> 4;
    const int colB = lane & 15;
    const int base = offsets[k];

    int tr = tile * 16 + colB;
    if (tr >= cnt) tr = cnt - 1;                 // pad: duplicate last row (benign rewrite)
    const int row = perm[base + tr];
    const int n0  = blockIdx.z * 512 + wave * 64;

    const ushort_t* aRow  = hbuf + ((size_t)k * BB + row) * HH;
    const ushort_t* wBase = w2t + (size_t)k * DD * HH;

    ABu a[8];
#pragma unroll
    for (int s = 0; s < 8; ++s) {
        int h0 = s * 32;
        a[s].h[0] = *(const v8bf*)(aRow + h0 + half * 16);
        a[s].h[1] = *(const v8bf*)(aRow + h0 + 8 + half * 16);
    }

    v8f acc[4] = {};
#pragma unroll
    for (int s = 0; s < 8; ++s) {
#pragma unroll
        for (int t = 0; t < 4; ++t) {
            int col = n0 + t * 16 + colB;
            v16bf b = *(const v16bf*)(wBase + (size_t)col * HH + s * 32 + half * 16);
            acc[t] = wmma_bf16(a[s].v, b, acc[t]);
        }
    }
#pragma unroll
    for (int t = 0; t < 4; ++t) {
        int col = n0 + t * 16 + colB;
        float bias = b2[k * DD + col];
#pragma unroll
        for (int r = 0; r < 8; ++r) {
            // Output row for C tile-row (r + half*8) is held as 'row' by that lane.
            int orow = __shfl(row, r + half * 8, 32);
            out[(size_t)orow * DD + col] = acc[t][r] + bias;
        }
    }
}

// ---------------------------------------------------------------------- launch

extern "C" void kernel_launch(void* const* d_in, const int* in_sizes, int n_in,
                              void* d_out, int out_size, void* d_ws, size_t ws_size,
                              hipStream_t stream) {
    const float* x  = (const float*)d_in[0];   // [B,D]
    const float* W1 = (const float*)d_in[1];   // [K,D,H]
    const float* b1 = (const float*)d_in[2];   // [K,H]
    const float* W2 = (const float*)d_in[3];   // [K,H,D]
    const float* b2 = (const float*)d_in[4];   // [K,D]
    float* out = (float*)d_out;

    char* ws = (char*)d_ws;
    size_t off = 0;
    ushort_t* xb   = (ushort_t*)(ws + off); off += (size_t)BB * DD * 2;       // 16 MB
    ushort_t* w1t  = (ushort_t*)(ws + off); off += (size_t)KK * HH * DD * 2;  // 4 MB
    ushort_t* w2t  = (ushort_t*)(ws + off); off += (size_t)KK * DD * HH * 2;  // 4 MB
    ushort_t* hbuf = (ushort_t*)(ws + off); off += (size_t)KK * BB * HH * 2;  // 32 MB
    float*    err  = (float*)   (ws + off); off += (size_t)KK * BB * 4;       // 256 KB
    int*  assignv  = (int*)     (ws + off); off += (size_t)BB * 4;
    int*  perm     = (int*)     (ws + off); off += (size_t)BB * 4;
    int*  counts   = (int*)     (ws + off); off += 64;
    int*  offsets  = (int*)     (ws + off); off += 64;
    int*  cursors  = (int*)     (ws + off); off += 64;
    (void)ws_size; (void)in_sizes; (void)n_in; (void)out_size;

    k_zero_counts<<<1, 32, 0, stream>>>(counts);
    k_cvt_x<<<(BB * DD) / 256, 256, 0, stream>>>(x, xb);
    k_prep_w<<<(KK * DD * HH) / 256, 256, 0, stream>>>(W1, W2, w1t, w2t);

    k_gemm1<<<dim3(BB / 32, KK), 128, 0, stream>>>(xb, w1t, b1, hbuf);
    k_gemm2_err<<<dim3(BB / 256, KK), 256, 0, stream>>>(hbuf, w2t, b2, x, err);

    k_assign<<<BB / 256, 256, 0, stream>>>(err, assignv, counts);
    k_scan<<<1, 32, 0, stream>>>(counts, offsets, cursors);
    k_scatter<<<BB / 256, 256, 0, stream>>>(assignv, cursors, perm);

    k_final<<<dim3(BB / 16, KK, DD / 512), 256, 0, stream>>>(
        hbuf, w2t, b2, counts, offsets, perm, out);
}
